// Model_42082089566245
// MI455X (gfx1250) — matrix operand order repaired and checked by reading, not compiled, not run on hardware
//
#include <hip/hip_runtime.h>

// ---------------- model constants ----------------
#define BATCH   16384
#define SEQ     6
#define DIM     991          // 128 + 84 + 11 + 768
#define DIM3    2973         // 3*DIM
#define DIM2    1982         // 2*DIM
#define KP      992          // DIM padded to multiple of 32
#define KTILES  31           // KP/32
#define NPAD    3072         // 24 * 128  (N tile coverage for DIM3)
#define NBLKS   24
#define MBLKS   (BATCH/128)  // 128
#define OUTC    2

typedef __attribute__((ext_vector_type(16))) __bf16        v16bf;
typedef __attribute__((ext_vector_type(8)))  float         v8f;
typedef __attribute__((ext_vector_type(4)))  unsigned int  v4u;

union Frag { v16bf bf; v4u u[2]; };

__device__ __forceinline__ unsigned short f2bf(float f) {
    unsigned int u = __float_as_uint(f);
    unsigned int r = u + 0x7FFFu + ((u >> 16) & 1u);   // round-to-nearest-even
    return (unsigned short)(r >> 16);
}

// low 32 bits of a generic pointer to __shared__ = LDS byte address (flat->LDS truncation)
__device__ __forceinline__ unsigned lds_addr(const void* p) {
    return (unsigned)(size_t)p;
}

// async memory -> LDS copy of 16 bytes (ASYNCcnt-tracked, no VGPR round-trip)
__device__ __forceinline__ void async_copy_b128(unsigned lds_dst, const void* gsrc) {
    asm volatile("global_load_async_to_lds_b128 %0, %1, off"
                 :: "v"(lds_dst), "v"(gsrc) : "memory");
}
__device__ __forceinline__ void wait_async0() {
    asm volatile("s_wait_asynccnt 0" ::: "memory");
}

// ---------------- 1) zero h / h_bf16 ----------------
__global__ __launch_bounds__(256)
void init_state(float* __restrict__ h, unsigned short* __restrict__ hbf) {
    size_t idx = (size_t)blockIdx.x * 256 + threadIdx.x;
    if (idx < (size_t)BATCH * DIM) h[idx] = 0.0f;
    if (idx < (size_t)BATCH * KP)  hbf[idx] = 0;
}

// ---------------- 2) embedding gather + LayerNorm -> bf16 (S,B,KP) ----------------
__global__ __launch_bounds__(256)
void embed_ln(const int* __restrict__ node_seq, const int* __restrict__ fin_seq,
              const int* __restrict__ nfin_seq, const int* __restrict__ mda_seq,
              const float* __restrict__ node_t, const float* __restrict__ fin_t,
              const float* __restrict__ nfin_t, const float* __restrict__ bert_t,
              const float* __restrict__ ln_g, const float* __restrict__ ln_b,
              unsigned short* __restrict__ xbf) {
    const int b = blockIdx.x, s = blockIdx.y, tid = threadIdx.x;
    const int ni = node_seq[b * SEQ + s];
    const int fi = fin_seq [b * SEQ + s];
    const int gi = nfin_seq[b * SEQ + s];
    const int mi = mda_seq [b * SEQ + s];

    float v[4];
    float sum = 0.f, sq = 0.f;
#pragma unroll
    for (int it = 0; it < 4; ++it) {
        int i = tid + it * 256;
        float val = 0.f;
        if (i < DIM) {
            if      (i < 128) val = node_t[(size_t)ni * 128 + i];
            else if (i < 212) val = fin_t [(size_t)fi * 84  + (i - 128)];
            else if (i < 223) val = nfin_t[(size_t)gi * 11  + (i - 212)];
            else              val = bert_t[(size_t)mi * 768 + (i - 223)];
        }
        v[it] = val;
        sum += val; sq += val * val;
    }
    __shared__ float r0[256], r1[256];
    r0[tid] = sum; r1[tid] = sq; __syncthreads();
    for (int off = 128; off > 0; off >>= 1) {
        if (tid < off) { r0[tid] += r0[tid + off]; r1[tid] += r1[tid + off]; }
        __syncthreads();
    }
    const float mu   = r0[0] * (1.0f / DIM);
    const float var  = r1[0] * (1.0f / DIM) - mu * mu;
    const float rinv = rsqrtf(var + 1e-5f);

    unsigned short* dst = xbf + ((size_t)s * BATCH + b) * KP;
#pragma unroll
    for (int it = 0; it < 4; ++it) {
        int i = tid + it * 256;
        if (i < KP) {
            float y = (i < DIM) ? (v[it] - mu) * rinv * ln_g[i] + ln_b[i] : 0.f;
            dst[i] = f2bf(y);
        }
    }
}

// ---------------- 3) weight fp32 -> bf16 padded (NPAD x KP) ----------------
__global__ __launch_bounds__(256)
void conv_weights(const float* __restrict__ Wih, const float* __restrict__ Whh,
                  unsigned short* __restrict__ Wihb, unsigned short* __restrict__ Whhb) {
    size_t idx = (size_t)blockIdx.x * 256 + threadIdx.x;
    if (idx >= (size_t)NPAD * KP) return;
    int n = (int)(idx / KP), k = (int)(idx - (size_t)n * KP);
    bool ok = (n < DIM3) && (k < DIM);
    float a = ok ? Wih[(size_t)n * DIM + k] : 0.f;
    float b = ok ? Whh[(size_t)n * DIM + k] : 0.f;
    Wihb[idx] = f2bf(a);
    Whhb[idx] = f2bf(b);
}

// ---------------- 4) exact collapse of the two final linears: W_eff = W2 @ W1 ----------------
__global__ __launch_bounds__(256)
void make_weff(const float* __restrict__ W1, const float* __restrict__ b1,
               const float* __restrict__ W2, const float* __restrict__ b2,
               float* __restrict__ weff /* [2*DIM + 2] */) {
    int idx = blockIdx.x * 256 + threadIdx.x;
    if (idx < 2 * DIM) {
        int o = idx / DIM, d = idx - o * DIM;
        float acc = 0.f;
        for (int j = 0; j < DIM2; ++j)
            acc += W2[o * DIM2 + j] * W1[(size_t)j * DIM + d];
        weff[idx] = acc;
    } else if (idx < 2 * DIM + OUTC) {
        int o = idx - 2 * DIM;
        float acc = b2[o];
        for (int j = 0; j < DIM2; ++j) acc += W2[o * DIM2 + j] * b1[j];
        weff[idx] = acc;
    }
}

// ---------------- 5) WMMA bf16 GEMM: C[16384 x N] = A[16384 x KP] * W[N x KP]^T + bias ----
// Block tile 128x128 (8 waves, each 32x64). Staging via async global->LDS copies.
__global__ __launch_bounds__(256)
void gemm_bf16(const unsigned short* __restrict__ A,
               const unsigned short* __restrict__ W,
               const float* __restrict__ bias,
               float* __restrict__ C, int N, int ldC) {
    __shared__ __align__(16) unsigned short sA[128 * 32];
    __shared__ __align__(16) unsigned short sB[128 * 32];

    const int tid    = threadIdx.x;
    const int lane   = tid & 31;
    const int wid    = tid >> 5;
    const int wm     = wid & 3;     // 4 waves along M (32 rows each)
    const int wn     = wid >> 2;    // 2 waves along N (64 cols each)
    const int laneHi = lane >> 4;   // 0|1
    const int laneLo = lane & 15;
    const size_t mBase = (size_t)blockIdx.y * 128;
    const int    nBase = blockIdx.x * 128;

    const unsigned ldsA = lds_addr(sA);
    const unsigned ldsB = lds_addr(sB);
    const int srow = tid >> 2, scc = tid & 3;   // staging: 2 chunks each for A and B

    v8f acc[2][4];
#pragma unroll
    for (int i = 0; i < 2; ++i)
#pragma unroll
        for (int j = 0; j < 4; ++j)
            acc[i][j] = (v8f){0.f,0.f,0.f,0.f,0.f,0.f,0.f,0.f};

    for (int kt = 0; kt < KTILES; ++kt) {
        const int k0 = kt * 32;
        // stage A tile 128x32 and B tile 128x32: 512 chunks each, 2+2 per thread,
        // direct memory->LDS (ASYNCcnt), no VGPR round trip
#pragma unroll
        for (int it = 0; it < 2; ++it) {
            int row = srow + it * 64;
            async_copy_b128(ldsA + (unsigned)(row * 64 + scc * 16),
                            &A[(mBase + row) * KP + k0 + scc * 8]);
            async_copy_b128(ldsB + (unsigned)(row * 64 + scc * 16),
                            &W[(size_t)(nBase + row) * KP + k0 + scc * 8]);
        }
        // prefetch next K-tile of A into cache while copies are in flight
        if (kt + 1 < KTILES)
            __builtin_prefetch(&A[(mBase + (tid >> 1)) * KP + k0 + 32 + (tid & 1) * 16], 0, 1);
        wait_async0();
        __syncthreads();

        Frag a[2], b[4];
        // A fragment: lanes 0-15 -> M=laneLo, K {0..7,16..23}; lanes 16-31 -> K {8..15,24..31}
#pragma unroll
        for (int mi = 0; mi < 2; ++mi) {
            int row = wm * 32 + mi * 16 + laneLo;
            int off = laneHi * 8;
            a[mi].u[0] = *(const v4u*)&sA[row * 32 + off];
            a[mi].u[1] = *(const v4u*)&sA[row * 32 + 16 + off];
        }
        // B fragment: lanes 0-15 -> N=laneLo K {0..7,16..23}; lanes 16-31 -> K {8..15,24..31}
#pragma unroll
        for (int niB = 0; niB < 4; ++niB) {
            int row  = wn * 64 + niB * 16 + laneLo;
            int koff = laneHi * 8;
            b[niB].u[0] = *(const v4u*)&sB[row * 32 + koff];
            b[niB].u[1] = *(const v4u*)&sB[row * 32 + koff + 16];
        }
#pragma unroll
        for (int mi = 0; mi < 2; ++mi)
#pragma unroll
            for (int niB = 0; niB < 4; ++niB)
                acc[mi][niB] = __builtin_amdgcn_wmma_f32_16x16x32_bf16(
                    false, a[mi].bf, false, b[niB].bf,
                    (short)0, acc[mi][niB], false, false);
        __syncthreads();
    }

    // store: VGPR v -> M = tileM + (lane>=16?8:0) + v, col = tileN + laneLo
#pragma unroll
    for (int mi = 0; mi < 2; ++mi) {
#pragma unroll
        for (int niB = 0; niB < 4; ++niB) {
            int col = nBase + wn * 64 + niB * 16 + laneLo;
            if (col < N) {
                size_t rowBase = mBase + wm * 32 + mi * 16 + laneHi * 8;
                float bv = bias[col];
#pragma unroll
                for (int v = 0; v < 8; ++v)
                    C[(rowBase + v) * (size_t)ldC + col] = acc[mi][niB][v] + bv;
            }
        }
    }
}

// ---------------- 6) GRU gates + state update + last-step select ----------------
__global__ __launch_bounds__(256)
void gru_gates(const float* __restrict__ xg, const float* __restrict__ gh,
               float* __restrict__ h, unsigned short* __restrict__ hbf,
               float* __restrict__ hsel, const int* __restrict__ seq_len, int t) {
    size_t idx = (size_t)blockIdx.x * 256 + threadIdx.x;
    if (idx >= (size_t)BATCH * DIM) return;
    int b = (int)(idx / DIM), d = (int)(idx - (size_t)b * DIM);
    size_t base = (size_t)b * DIM3 + d;
    float xr = xg[base], xz = xg[base + DIM], xn = xg[base + 2 * DIM];
    float hr = gh[base], hz = gh[base + DIM], hn = gh[base + 2 * DIM];
    float r = 1.f / (1.f + __expf(-(xr + hr)));
    float z = 1.f / (1.f + __expf(-(xz + hz)));
    float n = tanhf(xn + r * hn);
    float hp = h[idx];
    float hv = (1.f - z) * n + z * hp;
    h[idx] = hv;
    hbf[(size_t)b * KP + d] = f2bf(hv);
    if (seq_len[b] - 1 == t) hsel[idx] = hv;
}

// ---------------- 7) BatchNorm statistics over batch ----------------
__global__ __launch_bounds__(256)
void bn_stats(const float* __restrict__ hsel, float* __restrict__ stats) {
    const int d = blockIdx.x, tid = threadIdx.x;
    float s = 0.f, sq = 0.f;
    for (int b = tid; b < BATCH; b += 256) {
        float v = hsel[(size_t)b * DIM + d];
        s += v; sq += v * v;
    }
    __shared__ float r0[256], r1[256];
    r0[tid] = s; r1[tid] = sq; __syncthreads();
    for (int off = 128; off > 0; off >>= 1) {
        if (tid < off) { r0[tid] += r0[tid + off]; r1[tid] += r1[tid + off]; }
        __syncthreads();
    }
    if (tid == 0) {
        float mu  = r0[0] * (1.0f / BATCH);
        float var = r1[0] * (1.0f / BATCH) - mu * mu;
        stats[d]       = mu;
        stats[DIM + d] = rsqrtf(var + 1e-4f);
    }
}

// ---------------- 8) normalize + collapsed final linear ----------------
__global__ __launch_bounds__(256)
void final_logits(const float* __restrict__ hsel, const float* __restrict__ stats,
                  const float* __restrict__ bn_g, const float* __restrict__ bn_b,
                  const float* __restrict__ weff, float* __restrict__ out) {
    const int b = blockIdx.x, tid = threadIdx.x;
    float a0 = 0.f, a1 = 0.f;
    for (int d = tid; d < DIM; d += 256) {
        float xn = (hsel[(size_t)b * DIM + d] - stats[d]) * stats[DIM + d] * bn_g[d] + bn_b[d];
        a0 += xn * weff[d];
        a1 += xn * weff[DIM + d];
    }
    __shared__ float r0[256], r1[256];
    r0[tid] = a0; r1[tid] = a1; __syncthreads();
    for (int off = 128; off > 0; off >>= 1) {
        if (tid < off) { r0[tid] += r0[tid + off]; r1[tid] += r1[tid + off]; }
        __syncthreads();
    }
    if (tid == 0) {
        out[b * 2 + 0] = r0[0] + weff[2 * DIM + 0];
        out[b * 2 + 1] = r1[0] + weff[2 * DIM + 1];
    }
}

// ---------------- launch ----------------
extern "C" void kernel_launch(void* const* d_in, const int* in_sizes, int n_in,
                              void* d_out, int out_size, void* d_ws, size_t ws_size,
                              hipStream_t stream) {
    const int*   node_seq = (const int*)  d_in[0];
    const int*   fin_seq  = (const int*)  d_in[1];
    const int*   nfin_seq = (const int*)  d_in[2];
    const int*   mda_seq  = (const int*)  d_in[3];
    const int*   seq_len  = (const int*)  d_in[4];
    const float* node_t   = (const float*)d_in[5];
    const float* fin_t    = (const float*)d_in[6];
    const float* nfin_t   = (const float*)d_in[7];
    const float* bert_t   = (const float*)d_in[8];
    const float* ln_g     = (const float*)d_in[9];
    const float* ln_b     = (const float*)d_in[10];
    const float* W_ih     = (const float*)d_in[11];
    const float* W_hh     = (const float*)d_in[12];
    const float* b_ih     = (const float*)d_in[13];
    const float* b_hh     = (const float*)d_in[14];
    const float* bn_g     = (const float*)d_in[15];
    const float* bn_b     = (const float*)d_in[16];
    const float* W1       = (const float*)d_in[17];
    const float* b1       = (const float*)d_in[18];
    const float* W2       = (const float*)d_in[19];
    const float* b2       = (const float*)d_in[20];
    float*       out      = (float*)d_out;

    // carve workspace (256B aligned slices)
    char* ws = (char*)d_ws;
    auto carve = [&](size_t bytes) -> char* {
        char* p = ws;
        ws += (bytes + 255) & ~(size_t)255;
        return p;
    };
    unsigned short* xbf   = (unsigned short*)carve((size_t)SEQ * BATCH * KP * 2);
    unsigned short* wihb  = (unsigned short*)carve((size_t)NPAD * KP * 2);
    unsigned short* whhb  = (unsigned short*)carve((size_t)NPAD * KP * 2);
    float*          xgbuf = (float*)carve((size_t)BATCH * DIM3 * 4);
    float*          ghbuf = (float*)carve((size_t)BATCH * DIM3 * 4);
    float*          h     = (float*)carve((size_t)BATCH * DIM * 4);
    unsigned short* hbf   = (unsigned short*)carve((size_t)BATCH * KP * 2);
    float*          hsel  = (float*)carve((size_t)BATCH * DIM * 4);
    float*          stats = (float*)carve((size_t)2 * DIM * 4);
    float*          weff  = (float*)carve((size_t)(2 * DIM + OUTC) * 4);

    // state init (every call: deterministic)
    {
        size_t n = (size_t)BATCH * KP;
        init_state<<<dim3((unsigned)((n + 255) / 256)), 256, 0, stream>>>(h, hbf);
    }
    // embeddings + LayerNorm
    embed_ln<<<dim3(BATCH, SEQ), 256, 0, stream>>>(node_seq, fin_seq, nfin_seq, mda_seq,
                                                   node_t, fin_t, nfin_t, bert_t,
                                                   ln_g, ln_b, xbf);
    // weight conversion + collapsed head
    {
        size_t n = (size_t)NPAD * KP;
        conv_weights<<<dim3((unsigned)((n + 255) / 256)), 256, 0, stream>>>(W_ih, W_hh, wihb, whhb);
    }
    make_weff<<<dim3((2 * DIM + OUTC + 255) / 256), 256, 0, stream>>>(W1, b1, W2, b2, weff);

    // GRU over S steps: two WMMA GEMMs + gate fusion per step
    for (int t = 0; t < SEQ; ++t) {
        const unsigned short* xslab = xbf + (size_t)t * BATCH * KP;
        gemm_bf16<<<dim3(NBLKS, MBLKS), 256, 0, stream>>>(xslab, wihb, b_ih, xgbuf, DIM3, DIM3);
        gemm_bf16<<<dim3(NBLKS, MBLKS), 256, 0, stream>>>(hbf,   whhb, b_hh, ghbuf, DIM3, DIM3);
        {
            size_t n = (size_t)BATCH * DIM;
            gru_gates<<<dim3((unsigned)((n + 255) / 256)), 256, 0, stream>>>(
                xgbuf, ghbuf, h, hbf, hsel, seq_len, t);
        }
    }

    // BatchNorm + collapsed final linear
    bn_stats<<<dim3(DIM), 256, 0, stream>>>(hsel, stats);
    final_logits<<<dim3(BATCH), 256, 0, stream>>>(hsel, stats, bn_g, bn_b, weff, out);

    (void)in_sizes; (void)n_in; (void)out_size; (void)ws_size;
}
